// NeuralImplicitAccessibilityField_87333864997046
// MI455X (gfx1250) — matrix operand order repaired and checked by reading, not compiled, not run on hardware
//
#include <hip/hip_runtime.h>
#include <hip/hip_bf16.h>

// ---------------------------------------------------------------------------
// MI455X (gfx1250) fused implementation of NeuralImplicitAccessibilityField.
// All GEMM-shaped math on v_wmma_f32_16x16x32_f16 (f16 in / f32 accumulate);
// elementwise (sqrt/min distance scan, softmax, SiLU/tanh) on wave32 VALU.
// ---------------------------------------------------------------------------

typedef __attribute__((ext_vector_type(16))) _Float16 v16h;
typedef __attribute__((ext_vector_type(8)))  _Float16 v8h;
typedef __attribute__((ext_vector_type(8)))  float    v8f;

#define WMMA16(a, b, c) \
  __builtin_amdgcn_wmma_f32_16x16x32_f16(false, (a), false, (b), (short)0, (c), false, false)

namespace {
constexpr int NPTS = 262144;
constexpr int MRES = 1024;
constexpr int NBLK = NPTS / 128;   // 2048 blocks, 128 points per block

// Byte offsets shared between the ws const image and the LDS copy of it.
constexpr int OFF_RES   = 0;                   // float4[1024]  residue xyz + radius
constexpr int OFF_KEYS  = 16384;               // f16 [128][64] keys (pre-scaled 1/8), a-major
constexpr int OFF_NKEYS = 32768;               // f16 [128][64] nftm keys (pre-scaled 1/8)
constexpr int OFF_VALT  = 49152;               // f16 [64][128] values^T   (n-major B)
constexpr int OFF_SLOTT = 65536;               // f16 [64][128] slots^T
constexpr int OFF_CTLT  = 81920;               // f16 [64][128] controller^T
constexpr int OFF_CW2T  = 98304;               // f16 [64][64]  cp_w2^T
constexpr int OFF_W1T   = 106496;              // f16 [64][128] sr_w1[0:128]^T
constexpr int OFF_CONST = 122880;              // float[1024] small consts
constexpr int CONST_BYTES = 126976;            // 124 KB const image (16B multiple)

// float indices inside the const block
constexpr int CW1 = 0;    // cp_w1 [3][64]
constexpr int CB1 = 192;  // cp_b1 [64]
constexpr int CB2 = 256;  // cp_b2 [64]
constexpr int SB1 = 320;  // sr_b1 [64]
constexpr int SW2 = 384;  // sr_w2 [64]
constexpr int W1L = 448;  // sr_w1 last row [64]
constexpr int SB2 = 512;  // sr_b2

// LDS-only regions (after the const image)
constexpr int OFF_STAGE = CONST_BYTES;          // f16: 8 waves x [16][128]
constexpr int OFF_Q     = OFF_STAGE + 32768;    // f16: 8 waves x [16][64]
constexpr int OFF_SDF   = OFF_Q + 16384;        // f32: 8 waves x 16
constexpr int OFF_LOSS  = OFF_SDF + 512;        // f32: 8
constexpr int SMEM_BYTES = OFF_LOSS + 32;       // ~176.6 KB (<320 KB/WGP)

constexpr int OFF_PART  = CONST_BYTES;          // ws: float[NBLK] loss partials
}  // namespace

// --- WMMA fragment loaders (layouts per CDNA5 ISA 7.12.2, wave32) -----------
// A 16x32 f16: lanes 0-15 row r hold K 0-7 & 16-23; lanes 16-31 hold K 8-15 & 24-31.
__device__ inline v16h load_A_frag(const _Float16* base, int lane, int rowStride, int kOff) {
  const int r  = lane & 15;
  const int kb = (lane & 16) ? 8 : 0;
  const _Float16* p = base + r * rowStride + kOff + kb;
  v8h lo = *(const v8h*)(p);
  v8h hi = *(const v8h*)(p + 16);
  v16h o;
#pragma unroll
  for (int i = 0; i < 8; ++i) { o[i] = lo[i]; o[8 + i] = hi[i]; }
  return o;
}
// B 32x16 f16 from B^T (n-major) storage: lane's column = lane&15,
// lanes 0-15 hold K 0-7 & 16-23; lanes 16-31 hold K 8-15 & 24-31 (matches A).
__device__ inline v16h load_B_frag(const _Float16* baseT, int lane, int rowStride,
                                   int nOff, int kOff) {
  const int c  = lane & 15;
  const int kb = (lane & 16) ? 8 : 0;
  const _Float16* p = baseT + (nOff + c) * rowStride + kOff + kb;
  v8h lo = *(const v8h*)(p);
  v8h hi = *(const v8h*)(p + 16);
  v16h o;
#pragma unroll
  for (int i = 0; i < 8; ++i) { o[i] = lo[i]; o[8 + i] = hi[i]; }
  return o;
}

// Row-wise softmax over 128 columns held as 8 C-tiles (16 cols per lane-half).
__device__ inline void softmax_rows(v8f L[8]) {
#pragma unroll
  for (int j = 0; j < 8; ++j) {
    float m = L[0][j];
#pragma unroll
    for (int nt = 1; nt < 8; ++nt) m = fmaxf(m, L[nt][j]);
#pragma unroll
    for (int d = 1; d < 16; d <<= 1) m = fmaxf(m, __shfl_xor(m, d, 32));
    float s = 0.f;
#pragma unroll
    for (int nt = 0; nt < 8; ++nt) { float e = __expf(L[nt][j] - m); L[nt][j] = e; s += e; }
#pragma unroll
    for (int d = 1; d < 16; d <<= 1) s += __shfl_xor(s, d, 32);
    const float inv = 1.f / s;
#pragma unroll
    for (int nt = 0; nt < 8; ++nt) L[nt][j] *= inv;
  }
}

__device__ inline float silu_f(float x) { return x / (1.f + __expf(-x)); }

// ---------------------------------------------------------------------------
// Prep kernel: anchor-side math (tiny) + pack all operands into ws const image.
// ---------------------------------------------------------------------------
__global__ __launch_bounds__(256) void prep_kernel(
    const float* __restrict__ rpos, const int* __restrict__ rtype,
    const float* __restrict__ sf, const float* __restrict__ anch,
    const float* __restrict__ rtab,
    const float* __restrict__ mw_w1, const float* __restrict__ mw_b1,
    const float* __restrict__ mw_w2, const float* __restrict__ mw_b2,
    const float* __restrict__ cp_w1, const float* __restrict__ cp_b1,
    const float* __restrict__ cp_w2, const float* __restrict__ cp_b2,
    const float* __restrict__ ak_w, const float* __restrict__ ak_b,
    const float* __restrict__ av_w, const float* __restrict__ av_b,
    const float* __restrict__ nkeys, const float* __restrict__ nslots,
    const float* __restrict__ sr_w1, const float* __restrict__ sr_b1,
    const float* __restrict__ sr_w2, const float* __restrict__ sr_b2,
    char* __restrict__ ws) {
  __shared__ float hid[128 * 64];
  const int tid = threadIdx.x;

  // residues: xyz + looked-up radius
  float4* resO = (float4*)(ws + OFF_RES);
  for (int m = tid; m < MRES; m += 256)
    resO[m] = make_float4(rpos[3 * m], rpos[3 * m + 1], rpos[3 * m + 2], rtab[rtype[m]]);

  // hidden = silu(mem_in @ mw_w1 + b1), mem_in = [sf | anch]
  for (int idx = tid; idx < 128 * 64; idx += 256) {
    const int a = idx >> 6, c = idx & 63;
    float acc = mw_b1[c];
    for (int k = 0; k < 64; ++k) acc += sf[a * 64 + k] * mw_w1[k * 64 + c];
    for (int k = 0; k < 16; ++k) acc += anch[a * 16 + k] * mw_w1[(64 + k) * 64 + c];
    hid[idx] = silu_f(acc);
  }
  __syncthreads();

  // controller^T (n-major)
  _Float16* ctlT = (_Float16*)(ws + OFF_CTLT);
  for (int idx = tid; idx < 64 * 128; idx += 256) {
    const int n = idx >> 7, a = idx & 127;
    float acc = mw_b2[n];
    for (int k = 0; k < 64; ++k) acc += hid[a * 64 + k] * mw_w2[k * 64 + n];
    ctlT[idx] = (_Float16)acc;
  }
  // keys (a-major, pre-scaled by 1/sqrt(MEM)=0.125)
  _Float16* keysO = (_Float16*)(ws + OFF_KEYS);
  for (int idx = tid; idx < 128 * 64; idx += 256) {
    const int a = idx >> 6, k = idx & 63;
    float acc = ak_b[k];
    for (int j = 0; j < 16; ++j) acc += anch[a * 16 + j] * ak_w[j * 64 + k];
    keysO[idx] = (_Float16)(acc * 0.125f);
  }
  // nftm keys (a-major, pre-scaled)
  _Float16* nkeysO = (_Float16*)(ws + OFF_NKEYS);
  for (int idx = tid; idx < 128 * 64; idx += 256)
    nkeysO[idx] = (_Float16)(nkeys[idx] * 0.125f);
  // values^T
  _Float16* valT = (_Float16*)(ws + OFF_VALT);
  for (int idx = tid; idx < 64 * 128; idx += 256) {
    const int n = idx >> 7, a = idx & 127;
    float acc = av_b[n];
    for (int k = 0; k < 64; ++k) acc += sf[a * 64 + k] * av_w[k * 64 + n];
    for (int k = 0; k < 16; ++k) acc += anch[a * 16 + k] * av_w[(64 + k) * 64 + n];
    valT[idx] = (_Float16)acc;
  }
  // slots^T
  _Float16* slotT = (_Float16*)(ws + OFF_SLOTT);
  for (int idx = tid; idx < 64 * 128; idx += 256) {
    const int n = idx >> 7, a = idx & 127;
    slotT[idx] = (_Float16)nslots[a * 64 + n];
  }
  // cp_w2^T
  _Float16* cw2T = (_Float16*)(ws + OFF_CW2T);
  for (int idx = tid; idx < 64 * 64; idx += 256) {
    const int n = idx >> 6, k = idx & 63;
    cw2T[idx] = (_Float16)cp_w2[k * 64 + n];
  }
  // sr_w1[0:128]^T
  _Float16* w1T = (_Float16*)(ws + OFF_W1T);
  for (int idx = tid; idx < 64 * 128; idx += 256) {
    const int n = idx >> 7, k = idx & 127;
    w1T[idx] = (_Float16)sr_w1[k * 64 + n];
  }
  // small f32 consts
  float* cb = (float*)(ws + OFF_CONST);
  for (int i = tid; i < 192; i += 256) cb[CW1 + i] = cp_w1[i];
  if (tid < 64) {
    cb[CB1 + tid] = cp_b1[tid];
    cb[CB2 + tid] = cp_b2[tid];
    cb[SB1 + tid] = sr_b1[tid];
    cb[SW2 + tid] = sr_w2[tid];
    cb[W1L + tid] = sr_w1[128 * 64 + tid];
  }
  if (tid == 0) cb[SB2] = sr_b2[0];
}

// ---------------------------------------------------------------------------
// Fused per-point kernel: 8 waves x 16 points per block.
// ---------------------------------------------------------------------------
__global__ __launch_bounds__(256) void fused_kernel(const float* __restrict__ coords,
                                                    const char* __restrict__ ws,
                                                    float* __restrict__ out,
                                                    float* __restrict__ partials) {
  extern __shared__ char smem[];
  const int tid = threadIdx.x;

  // Pull 124 KB const image into LDS (L2-resident source, 16B vector copies).
  {
    const uint4* src = (const uint4*)ws;
    uint4* dst = (uint4*)smem;
    for (int i = tid; i < CONST_BYTES / 16; i += 256) dst[i] = src[i];
  }
  __syncthreads();

  const int wave = tid >> 5, lane = tid & 31;
  const int p16 = lane & 15, hb = lane >> 4;   // hb = lane half (0/1)

  const float4*    sRes   = (const float4*)(smem + OFF_RES);
  const _Float16*  sKeys  = (const _Float16*)(smem + OFF_KEYS);
  const _Float16*  sNKeys = (const _Float16*)(smem + OFF_NKEYS);
  const _Float16*  sValT  = (const _Float16*)(smem + OFF_VALT);
  const _Float16*  sSlotT = (const _Float16*)(smem + OFF_SLOTT);
  const _Float16*  sCtlT  = (const _Float16*)(smem + OFF_CTLT);
  const _Float16*  sCw2T  = (const _Float16*)(smem + OFF_CW2T);
  const _Float16*  sW1T   = (const _Float16*)(smem + OFF_W1T);
  const float*     sC     = (const float*)(smem + OFF_CONST);
  _Float16* sStage = (_Float16*)(smem + OFF_STAGE) + wave * 16 * 128;
  _Float16* sQ     = (_Float16*)(smem + OFF_Q) + wave * 16 * 64;
  float*    sSdf   = (float*)(smem + OFF_SDF) + wave * 16;
  float*    sLoss  = (float*)(smem + OFF_LOSS);

  const int r0 = blockIdx.x * 128 + wave * 16;   // first point row of this wave
  const int prow = r0 + p16;
  const float x = coords[3 * prow], y = coords[3 * prow + 1], z = coords[3 * prow + 2];

  // --- Phase 1: coord MLP hidden (silu) + residue distance min-reduce -------
  {
    const int j0 = hb * 32;                      // each lane-half does 32 of 64 dims
#pragma unroll 8
    for (int j = 0; j < 32; ++j) {
      const int c = j0 + j;
      float h = sC[CB1 + c] + x * sC[CW1 + c] + y * sC[CW1 + 64 + c] + z * sC[CW1 + 128 + c];
      sStage[p16 * 128 + c] = (_Float16)silu_f(h);
    }
    float mn = 3.4e38f;
    const int m0 = hb * 512;                     // halves of M split across lane halves
#pragma unroll 4
    for (int m = m0; m < m0 + 512; ++m) {
      const float4 R = sRes[m];
      const float dx = x - R.x, dy = y - R.y, dz = z - R.z;
      mn = fminf(mn, __builtin_sqrtf(dx * dx + dy * dy + dz * dz) - R.w);
    }
    mn = fminf(mn, __shfl_xor(mn, 16, 32));
    sSdf[p16] = mn;
  }
  __syncthreads();

  // --- Phase 2: q = hidden @ cp_w2 + b2 (WMMA), stage q as f16 --------------
  {
    v8f qc[4];
#pragma unroll
    for (int nt = 0; nt < 4; ++nt) {
      const float b = sC[CB2 + nt * 16 + p16];
      v8f c = {b, b, b, b, b, b, b, b};
#pragma unroll
      for (int kc = 0; kc < 2; ++kc) {
        v16h a  = load_A_frag(sStage, lane, 128, kc * 32);
        v16h bb = load_B_frag(sCw2T, lane, 64, nt * 16, kc * 32);
        c = WMMA16(a, bb, c);
      }
      qc[nt] = c;
    }
#pragma unroll
    for (int nt = 0; nt < 4; ++nt)
#pragma unroll
      for (int j = 0; j < 8; ++j)
        sQ[(j + hb * 8) * 64 + nt * 16 + p16] = (_Float16)qc[nt][j];
  }
  __syncthreads();

  v8f L[8];

  // --- Phase 3: logits1 = q @ keys^T (pre-scaled), softmax, stage attn ------
#pragma unroll
  for (int nt = 0; nt < 8; ++nt) {
    v8f c = {};
#pragma unroll
    for (int kc = 0; kc < 2; ++kc) {
      v16h a  = load_A_frag(sQ, lane, 64, kc * 32);
      v16h bb = load_B_frag(sKeys, lane, 64, nt * 16, kc * 32);
      c = WMMA16(a, bb, c);
    }
    L[nt] = c;
  }
  softmax_rows(L);
#pragma unroll
  for (int nt = 0; nt < 8; ++nt)
#pragma unroll
    for (int j = 0; j < 8; ++j)
      sStage[(j + hb * 8) * 128 + nt * 16 + p16] = (_Float16)L[nt][j];
  __syncthreads();

  // --- Phase 4: ctx = attn @ values ----------------------------------------
  v8f ctx[4];
  {
    v16h ar[4];
#pragma unroll
    for (int kc = 0; kc < 4; ++kc) ar[kc] = load_A_frag(sStage, lane, 128, kc * 32);
#pragma unroll
    for (int nt = 0; nt < 4; ++nt) {
      v8f c = {};
#pragma unroll
      for (int kc = 0; kc < 4; ++kc)
        c = WMMA16(ar[kc], load_B_frag(sValT, lane, 128, nt * 16, kc * 32), c);
      ctx[nt] = c;
    }
  }
  __syncthreads();

  // --- Phase 5: logits2 = q @ nftm_keys^T, softmax -> read_w (output+stage) -
#pragma unroll
  for (int nt = 0; nt < 8; ++nt) {
    v8f c = {};
#pragma unroll
    for (int kc = 0; kc < 2; ++kc) {
      v16h a  = load_A_frag(sQ, lane, 64, kc * 32);
      v16h bb = load_B_frag(sNKeys, lane, 64, nt * 16, kc * 32);
      c = WMMA16(a, bb, c);
    }
    L[nt] = c;
  }
  softmax_rows(L);
  {
    float* orw = out + (size_t)2 * NPTS + 1;
#pragma unroll
    for (int nt = 0; nt < 8; ++nt)
#pragma unroll
      for (int j = 0; j < 8; ++j) {
        const int row = j + hb * 8, col = nt * 16 + p16;
        orw[(size_t)(r0 + row) * 128 + col] = L[nt][j];
        sStage[row * 128 + col] = (_Float16)L[nt][j];
      }
  }
  __syncthreads();

  // --- Phase 6: memory_context = rw @ slots; ctx = 0.5*(ctx + rw @ ctl) -----
  v8f mc[4];
  {
    v16h ar[4];
#pragma unroll
    for (int kc = 0; kc < 4; ++kc) ar[kc] = load_A_frag(sStage, lane, 128, kc * 32);
#pragma unroll
    for (int nt = 0; nt < 4; ++nt) {
      v8f c = {};
#pragma unroll
      for (int kc = 0; kc < 4; ++kc)
        c = WMMA16(ar[kc], load_B_frag(sSlotT, lane, 128, nt * 16, kc * 32), c);
      mc[nt] = c;
    }
#pragma unroll
    for (int nt = 0; nt < 4; ++nt) {
      v8f c = ctx[nt];                            // feed ctx in as WMMA C operand
#pragma unroll
      for (int kc = 0; kc < 4; ++kc)
        c = WMMA16(ar[kc], load_B_frag(sCtlT, lane, 128, nt * 16, kc * 32), c);
      ctx[nt] = c * 0.5f;
    }
  }
  __syncthreads();

  // --- Phase 7: emit context, stage feat = [ctx | mc] as f16 ----------------
  {
    float* octx = out + (size_t)2 * NPTS + 1 + (size_t)NPTS * 128;
#pragma unroll
    for (int nt = 0; nt < 4; ++nt)
#pragma unroll
      for (int j = 0; j < 8; ++j) {
        const int row = j + hb * 8, col = nt * 16 + p16;
        octx[(size_t)(r0 + row) * 64 + col] = ctx[nt][j];
        sStage[row * 128 + col]       = (_Float16)ctx[nt][j];
        sStage[row * 128 + 64 + col]  = (_Float16)mc[nt][j];
      }
  }
  __syncthreads();

  // --- Phase 8: sdf residual MLP + outputs ----------------------------------
  {
    v16h af[4];
#pragma unroll
    for (int kc = 0; kc < 4; ++kc) af[kc] = load_A_frag(sStage, lane, 128, kc * 32);
    v8f hh[4];
#pragma unroll
    for (int nt = 0; nt < 4; ++nt) {
      const float b = sC[SB1 + nt * 16 + p16];
      v8f c = {b, b, b, b, b, b, b, b};
#pragma unroll
      for (int kc = 0; kc < 4; ++kc)
        c = WMMA16(af[kc], load_B_frag(sW1T, lane, 128, nt * 16, kc * 32), c);
      hh[nt] = c;
    }
    float lacc = 0.f;
#pragma unroll
    for (int j = 0; j < 8; ++j) {
      const int row = j + hb * 8;
      const float sb = sSdf[row];
      float local = 0.f;
#pragma unroll
      for (int nt = 0; nt < 4; ++nt) {
        const int col = nt * 16 + p16;
        float v = hh[nt][j] + sb * sC[W1L + col];   // K=129th rank-1 term
        local += silu_f(v) * sC[SW2 + col];
      }
#pragma unroll
      for (int d = 1; d < 16; d <<= 1) local += __shfl_xor(local, d, 32);
      const float sd = sb + 0.25f * tanhf(local + sC[SB2]);
      if (p16 == 0) {
        const int grow = r0 + row;
        out[grow] = sd;
        out[NPTS + grow] = 1.f / (1.f + __expf(-8.f * sd));
        lacc += fmaxf(-sd, 0.f);
      }
    }
    lacc += __shfl_xor(lacc, 16, 32);
    if (lane == 0) sLoss[wave] = lacc;
  }
  __syncthreads();
  if (tid == 0) {
    float t = 0.f;
    for (int w = 0; w < 8; ++w) t += sLoss[w];
    partials[blockIdx.x] = t;
  }
}

// Deterministic fixed-order loss reduction.
__global__ __launch_bounds__(256) void reduce_kernel(const float* __restrict__ partials,
                                                     float* __restrict__ out) {
  __shared__ float sm[256];
  float s = 0.f;
  for (int i = threadIdx.x; i < NBLK; i += 256) s += partials[i];
  sm[threadIdx.x] = s;
  __syncthreads();
  for (int st = 128; st > 0; st >>= 1) {
    if (threadIdx.x < st) sm[threadIdx.x] += sm[threadIdx.x + st];
    __syncthreads();
  }
  if (threadIdx.x == 0) out[(size_t)2 * NPTS] = sm[0] / (float)NPTS;
}

extern "C" void kernel_launch(void* const* d_in, const int* in_sizes, int n_in,
                              void* d_out, int out_size, void* d_ws, size_t ws_size,
                              hipStream_t stream) {
  const float* coords = (const float*)d_in[0];
  const float* rpos   = (const float*)d_in[1];
  const int*   rtype  = (const int*)d_in[2];
  const float* sf     = (const float*)d_in[3];
  const float* anch   = (const float*)d_in[4];
  const float* rtab   = (const float*)d_in[5];
  const float* mw_w1  = (const float*)d_in[6];
  const float* mw_b1  = (const float*)d_in[7];
  const float* mw_w2  = (const float*)d_in[8];
  const float* mw_b2  = (const float*)d_in[9];
  const float* cp_w1  = (const float*)d_in[10];
  const float* cp_b1  = (const float*)d_in[11];
  const float* cp_w2  = (const float*)d_in[12];
  const float* cp_b2  = (const float*)d_in[13];
  const float* ak_w   = (const float*)d_in[14];
  const float* ak_b   = (const float*)d_in[15];
  const float* av_w   = (const float*)d_in[16];
  const float* av_b   = (const float*)d_in[17];
  const float* nkeys  = (const float*)d_in[18];
  const float* nslots = (const float*)d_in[19];
  const float* sr_w1  = (const float*)d_in[20];
  const float* sr_b1  = (const float*)d_in[21];
  const float* sr_w2  = (const float*)d_in[22];
  const float* sr_b2  = (const float*)d_in[23];
  char*  ws  = (char*)d_ws;
  float* out = (float*)d_out;

  (void)hipFuncSetAttribute(reinterpret_cast<const void*>(fused_kernel),
                            hipFuncAttributeMaxDynamicSharedMemorySize, SMEM_BYTES);

  prep_kernel<<<1, 256, 0, stream>>>(rpos, rtype, sf, anch, rtab,
                                     mw_w1, mw_b1, mw_w2, mw_b2,
                                     cp_w1, cp_b1, cp_w2, cp_b2,
                                     ak_w, ak_b, av_w, av_b,
                                     nkeys, nslots, sr_w1, sr_b1, sr_w2, sr_b2, ws);
  fused_kernel<<<NBLK, 256, SMEM_BYTES, stream>>>(coords, ws, out,
                                                  (float*)(ws + OFF_PART));
  reduce_kernel<<<1, 256, 0, stream>>>((const float*)(ws + OFF_PART), out);
}
